// LSTM_35510789603866
// MI455X (gfx1250) — matrix operand order repaired and checked by reading, not compiled, not run on hardware
//
#include <hip/hip_runtime.h>
#include <hip/hip_bf16.h>
#include <math.h>

typedef __attribute__((ext_vector_type(16))) _Float16      v16h;
typedef __attribute__((ext_vector_type(8)))  float         v8f;
typedef __attribute__((ext_vector_type(4)))  unsigned int  v4u;
typedef __attribute__((ext_vector_type(8)))  int           v8i;
typedef __attribute__((ext_vector_type(4)))  int           v4i;

// Problem constants (from reference)
constexpr int S    = 262144;
constexpr int I    = 2;
constexpr int H    = 100;
constexpr int G    = 4 * H;     // 400 gate rows
constexpr int NT   = 25;        // row tiles of 16 (400/16)
constexpr int NKC  = 4;         // K chunks of 32 (K=100 padded to 128)
constexpr int KP   = 128;       // padded K
constexpr int CH   = 2048;      // timesteps per staged input chunk (2^11)
constexpr int CHSH = 11;        // log2(CH)
constexpr int BLK  = 256;       // 8 wave32 waves on one WGP
constexpr int SLICE_DW = CH * I / 8;  // dwords per wave's TDM slice (512)

// 1D Tensor-Data-Mover load: ndw dwords from gaddr -> LDS laddr (TENSORcnt).
// D# layout per CDNA5 ISA.
__device__ __forceinline__ void tdm_load_1d(unsigned long long gaddr,
                                            unsigned int laddr,
                                            unsigned int ndw) {
    v4u g0;
    g0[0] = 1u;                                   // count=1, load, no gather
    g0[1] = laddr;                                // lds_addr (bytes)
    g0[2] = (unsigned int)gaddr;                  // global_addr[31:0]
    g0[3] = (unsigned int)((gaddr >> 32) & 0x01FFFFFFu) | (2u << 30); // type=2
    v8i g1;
    g1[0] = (int)(2u << 16);                      // data_size=2 (4 bytes)
    g1[1] = (int)((ndw & 0xFFFFu) << 16);         // tensor_dim0[15:0] (abar=0)
    g1[2] = (int)((ndw >> 16) | (1u << 16));      // tensor_dim0[31:16], dim1=1
    g1[3] = (int)((ndw & 0xFFFFu) << 16);         // tile_dim0
    g1[4] = 1;                                    // tile_dim1=1, tile_dim2=0
    g1[5] = (int)ndw;                             // tensor_dim0_stride[31:0]
    g1[6] = 0;                                    // stride0 hi / stride1 lo
    g1[7] = 0;                                    // stride1 hi
    v4i z4 = {0, 0, 0, 0};
#if __has_include(<hip/amd_detail/amd_gfx1250_TDM.h>)
    v8i z8 = {0, 0, 0, 0, 0, 0, 0, 0};            // clang-23 / therock: 6-arg
    __builtin_amdgcn_tensor_load_to_lds(g0, g1, z4, z4, z8, 0);
#else
    __builtin_amdgcn_tensor_load_to_lds(g0, g1, z4, z4, 0);  // ROCm 7.2: 5-arg
#endif
}

__launch_bounds__(BLK, 1)
__global__ void lstm_cdna5_kernel(const float* __restrict__ input_seq,
                                  const float* __restrict__ h0,
                                  const float* __restrict__ c0,
                                  const float* __restrict__ W_ih,
                                  const float* __restrict__ W_hh,
                                  const float* __restrict__ b_ih,
                                  const float* __restrict__ b_hh,
                                  const float* __restrict__ W_out,
                                  const float* __restrict__ b_out,
                                  float* __restrict__ out)
{
    __shared__ __align__(32) _Float16 h16[KP];          // h_{t-1}, f16, padded
    __shared__ __align__(16) float    partial[G];       // W_hh @ h results
    __shared__ __align__(16) float    xbuf[2][CH * I];  // double-buffered input

    const int tid    = threadIdx.x;
    const int wave   = tid >> 5;
    const int lane   = tid & 31;
    const int m      = lane & 15;         // row-in-tile / N column id
    const int hiHalf = (lane >> 4) & 1;   // lane group 0-15 vs 16-31

    // ---------------- Preamble: W_hh -> resident WMMA A-fragments ----------
    // A 16x32, this code's contiguous K order: lanes 0-15 hold M=lane; element e -> K=e;
    // lanes 16-31 add K+=16 (K-contiguous, matching B).
    v16h afrag[4][NKC];
    #pragma unroll
    for (int i = 0; i < 4; ++i) {
        const int t = wave + 8 * i;           // row tile owned by this wave
        #pragma unroll
        for (int c = 0; c < NKC; ++c) {
            v16h a;
            #pragma unroll
            for (int e = 0; e < 16; ++e) {
                const int k = 32 * c + hiHalf * 16 + e;
                float v = 0.0f;
                if (t < NT && k < H) {
                    const int row = t * 16 + m;
                    v = W_hh[row * H + k];
                }
                a[e] = (_Float16)v;
            }
            afrag[i][c] = a;
        }
    }

    // Per-hidden-unit recurrent state + input projection constants in VGPRs
    float wih0[4] = {0, 0, 0, 0}, wih1[4] = {0, 0, 0, 0}, bias[4] = {0, 0, 0, 0};
    float wout = 0.0f, cst = 0.0f, hj = 0.0f;
    if (tid < H) {
        #pragma unroll
        for (int g = 0; g < 4; ++g) {
            const int r = g * H + tid;
            wih0[g] = W_ih[r * I + 0];
            wih1[g] = W_ih[r * I + 1];
            bias[g] = b_ih[r] + b_hh[r];
        }
        wout = W_out[tid];
        cst  = c0[tid];
        hj   = h0[tid];
        h16[tid] = (_Float16)hj;
    }
    if (tid >= H && tid < KP) h16[tid] = (_Float16)0.0f;

    // Kick off chunk 0 via the Tensor Data Mover: each wave DMAs its 2KB slice.
    const unsigned int mywave    = (unsigned int)__builtin_amdgcn_readfirstlane(wave);
    const unsigned int xbuf_lds  = (unsigned int)(uintptr_t)&xbuf[0][0];
    const unsigned long long gin = (unsigned long long)(uintptr_t)input_seq;
    tdm_load_1d(gin + mywave * (SLICE_DW * 4ull),
                xbuf_lds + mywave * (SLICE_DW * 4u), SLICE_DW);
    __syncthreads();

    // ---------------- Sequential recurrence ---------------------------------
    for (int t = 0; t < S; ++t) {
        if ((t & (CH - 1)) == 0) {
            const int buf = (t >> CHSH) & 1;
            // My slice of buffer `buf` has landed; then make it block-visible.
            __builtin_amdgcn_s_wait_tensorcnt(0);
            __syncthreads();
            // Prefill the other buffer with the next chunk while we compute.
            if (t + CH < S) {
                const unsigned long long ga =
                    gin + (unsigned long long)(t + CH) * I * 4ull +
                    mywave * (SLICE_DW * 4ull);
                const unsigned int la =
                    xbuf_lds + (unsigned int)(buf ^ 1) * (CH * I * 4u) +
                    mywave * (SLICE_DW * 4u);
                tdm_load_1d(ga, la, SLICE_DW);
                // Warm L2 two chunks ahead (global_prefetch_b8).
                if (t + 2 * CH < S && tid * 128 < CH * I * 4) {
                    const char* nxt =
                        (const char*)input_seq + (size_t)(t + 2 * CH) * I * 4;
                    __builtin_prefetch(nxt + tid * 128, 0, 1);
                }
            }
        }

        // B fragments: h replicated across all 16 N columns.
        // B 32x16, this code's contiguous K order: lane group selects K half; per-lane the 16
        // halves are K-contiguous -> one aligned 32B LDS read per chunk.
        v16h bfrag[NKC];
        #pragma unroll
        for (int c = 0; c < NKC; ++c)
            bfrag[c] = *(const v16h*)&h16[32 * c + hiHalf * 16];

        #pragma unroll
        for (int i = 0; i < 4; ++i) {
            const int tt = wave + 8 * i;
            if (tt < NT) {                      // wave-uniform: EXEC stays full
                v8f acc = {};
                #pragma unroll
                for (int c = 0; c < NKC; ++c)
                    acc = __builtin_amdgcn_wmma_f32_16x16x32_f16(
                        false, afrag[i][c], false, bfrag[c],
                        (short)0, acc, false, false);
                if (m == 0) {                   // column 0 holds the dot result
                    const int rb = tt * 16 + hiHalf * 8;
                    *(float4*)&partial[rb]     = make_float4(acc[0], acc[1], acc[2], acc[3]);
                    *(float4*)&partial[rb + 4] = make_float4(acc[4], acc[5], acc[6], acc[7]);
                }
            }
        }
        __syncthreads();

        // Gate nonlinearity + state update: one lane per hidden unit.
        if (tid < H) {
            const float* xb = &xbuf[(t >> CHSH) & 1][0];
            const int   ti = (t & (CH - 1)) * I;
            const float x0 = xb[ti + 0];
            const float x1 = xb[ti + 1];
            float gi = partial[tid]         + bias[0] + wih0[0] * x0 + wih1[0] * x1;
            float gf = partial[H + tid]     + bias[1] + wih0[1] * x0 + wih1[1] * x1;
            float gg = partial[2 * H + tid] + bias[2] + wih0[2] * x0 + wih1[2] * x1;
            float go = partial[3 * H + tid] + bias[3] + wih0[3] * x0 + wih1[3] * x1;
            gi = 1.0f / (1.0f + __expf(-gi));
            gf = 1.0f / (1.0f + __expf(-gf));
            gg = tanhf(gg);
            go = 1.0f / (1.0f + __expf(-go));
            cst = gf * cst + gi * gg;
            hj  = go * tanhf(cst);
            h16[tid] = (_Float16)hj;
        }
        __syncthreads();
    }

    // ---------------- Output head: out = h_T . W_out + b_out ---------------
    if (tid < H) partial[tid] = hj * wout;
    __syncthreads();
    if (tid == 0) {
        float s = b_out[0];
        #pragma unroll 4
        for (int j = 0; j < H; ++j) s += partial[j];
        out[0] = s;
    }
}

extern "C" void kernel_launch(void* const* d_in, const int* in_sizes, int n_in,
                              void* d_out, int out_size, void* d_ws, size_t ws_size,
                              hipStream_t stream) {
    (void)in_sizes; (void)n_in; (void)out_size; (void)d_ws; (void)ws_size;
    const float* input_seq = (const float*)d_in[0];
    const float* h0        = (const float*)d_in[1];
    const float* c0        = (const float*)d_in[2];
    const float* W_ih      = (const float*)d_in[3];
    const float* W_hh      = (const float*)d_in[4];
    const float* b_ih      = (const float*)d_in[5];
    const float* b_hh      = (const float*)d_in[6];
    const float* W_out     = (const float*)d_in[7];
    const float* b_out     = (const float*)d_in[8];
    float* out = (float*)d_out;

    // Single persistent workgroup: the LSTM chain is strictly sequential.
    lstm_cdna5_kernel<<<1, BLK, 0, stream>>>(input_seq, h0, c0, W_ih, W_hh,
                                             b_ih, b_hh, W_out, b_out, out);
}
